// mLSTMBlock_6030134084205
// MI455X (gfx1250) — matrix operand order repaired and checked by reading, not compiled, not run on hardware
//
#include <hip/hip_runtime.h>
#include <hip/hip_bf16.h>

typedef __attribute__((ext_vector_type(16))) _Float16 v16h;
typedef __attribute__((ext_vector_type(8)))  _Float16 v8h;
typedef __attribute__((ext_vector_type(8)))  float    v8f;
typedef __attribute__((ext_vector_type(4)))  float    v4f;

#define WMMA_F16(A, Bm, C) \
  __builtin_amdgcn_wmma_f32_16x16x32_f16(false, (A), false, (Bm), (short)0, (C), false, false)

#if defined(__HIP_DEVICE_COMPILE__) &&                                   \
    __has_builtin(__builtin_amdgcn_global_load_async_to_lds_b128) &&     \
    __has_builtin(__builtin_amdgcn_s_wait_asynccnt)
#define USE_ASYNC_LDS 1
#else
#define USE_ASYNC_LDS 0
#endif

// The async-to-LDS builtins take pointers to 16-byte int vectors
// (vector_size(16) int), with global-src / LDS-dst address spaces.
typedef int v4i_t __attribute__((vector_size(16)));
typedef __attribute__((address_space(1))) v4i_t as1_v4i;
typedef __attribute__((address_space(3))) v4i_t as3_v4i;

// On amdgcn a generic LDS address carries the LDS offset in its low 32 bits,
// and generic global addresses equal global addresses numerically.
__device__ __forceinline__ as3_v4i* to_lds_v4(const void* p) {
  return (as3_v4i*)(unsigned int)(uintptr_t)p;
}
__device__ __forceinline__ as1_v4i* to_glob_v4(const void* p) {
  return (as1_v4i*)(uintptr_t)p;
}

constexpr int B_  = 2;
constexpr int S_  = 2048;
constexpr int DM_ = 1024;
constexpr int H_  = 8;
constexpr int DI_ = 2048;
constexpr int DH_ = 256;
constexpr int K3_ = 3 * DI_;    // 6144
constexpr int MTOT_ = B_ * S_;  // 4096

// ---- fragment helpers (16-bit A-matrix 16x32 lane layout, ISA 7.12.2) ----
// lane&15 = M row; lane>>4 selects K-half: elems 0..7 -> K=kb..kb+7, 8..15 -> K=16+kb..
__device__ __forceinline__ v16h load_afrag(const _Float16* rowptr, int kb) {
  v8h lo = *(const v8h*)(rowptr + kb);
  v8h hi = *(const v8h*)(rowptr + 16 + kb);
  v16h r;
#pragma unroll
  for (int i = 0; i < 8; ++i) { r[i] = lo[i]; r[i + 8] = hi[i]; }
  return r;
}

__device__ __forceinline__ v8f zero8() {
  v8f z;
#pragma unroll
  for (int e = 0; e < 8; ++e) z[e] = 0.f;
  return z;
}

// =====================================================================
// Kernel 1: qkv = x @ W_qkv^T   (M=4096, N=6144, K=1024) -> f16 out
// double-buffered LDS, global loads of tile k+1 overlap WMMAs of tile k
// =====================================================================
__global__ __launch_bounds__(256) void gemm_qkv_kernel(
    const float* __restrict__ x, const float* __restrict__ Wqkv,
    _Float16* __restrict__ qkv) {
  __shared__ _Float16 As[2][128][32];
  __shared__ _Float16 Bs[2][32][128];

  const int tid  = threadIdx.x;
  const int lane = tid & 31;
  const int w    = tid >> 5;
  const int m0   = blockIdx.y * 128;
  const int n0   = blockIdx.x * 128;
  const int wm   = (w >> 1) * 32;  // wave M offset (4 waves down)
  const int wn   = (w & 1) * 64;   // wave N offset (2 waves across)

  v8f acc[2][4];
#pragma unroll
  for (int i = 0; i < 2; ++i)
#pragma unroll
    for (int j = 0; j < 4; ++j) acc[i][j] = zero8();

  const int ar = tid >> 1;         // tile row handled by this thread (0..127)
  const int ac = (tid & 1) * 16;   // k-col base (0 or 16)
  const float* gaBase = x    + (size_t)(m0 + ar) * DM_ + ac;
  const float* gbBase = Wqkv + (size_t)(n0 + ar) * DM_ + ac;

  v4f areg[4], breg[4];
  // prologue: stage tile 0 into buffer 0
#pragma unroll
  for (int j = 0; j < 4; ++j) {
    areg[j] = *(const v4f*)(gaBase + 4 * j);
    breg[j] = *(const v4f*)(gbBase + 4 * j);
  }
#pragma unroll
  for (int j = 0; j < 4; ++j)
#pragma unroll
    for (int e = 0; e < 4; ++e) {
      As[0][ar][ac + 4 * j + e] = (_Float16)areg[j][e];
      Bs[0][ac + 4 * j + e][ar] = (_Float16)breg[j][e];
    }
  __syncthreads();

  const int NT = DM_ / 32;
  for (int kt = 0; kt < NT; ++kt) {
    const int p = kt & 1;
    const bool more = (kt + 1 < NT);
    if (more) {
      const float* ga = gaBase + (size_t)(kt + 1) * 32;
      const float* gb = gbBase + (size_t)(kt + 1) * 32;
#pragma unroll
      for (int j = 0; j < 4; ++j) {
        areg[j] = *(const v4f*)(ga + 4 * j);
        breg[j] = *(const v4f*)(gb + 4 * j);
      }
      if (kt + 2 < NT) {
        __builtin_prefetch(gaBase + (size_t)(kt + 2) * 32, 0, 1);
        __builtin_prefetch(gbBase + (size_t)(kt + 2) * 32, 0, 1);
      }
    }

    v16h a[2], bf[4];
#pragma unroll
    for (int i = 0; i < 2; ++i)
      a[i] = load_afrag(&As[p][wm + i * 16 + (lane & 15)][0], (lane >> 4) * 8);
#pragma unroll
    for (int j = 0; j < 4; ++j)
#pragma unroll
      for (int s = 0; s < 16; ++s)
        bf[j][s] = Bs[p][(s < 8 ? s : s + 8) + (lane >> 4) * 8][wn + j * 16 + (lane & 15)];
#pragma unroll
    for (int i = 0; i < 2; ++i)
#pragma unroll
      for (int j = 0; j < 4; ++j) acc[i][j] = WMMA_F16(a[i], bf[j], acc[i][j]);

    if (more) {
#pragma unroll
      for (int j = 0; j < 4; ++j)
#pragma unroll
        for (int e = 0; e < 4; ++e) {
          As[p ^ 1][ar][ac + 4 * j + e] = (_Float16)areg[j][e];
          Bs[p ^ 1][ac + 4 * j + e][ar] = (_Float16)breg[j][e];
        }
    }
    __syncthreads();
  }

  // store (C layout: lanes0-15 N=lane,M=r ; lanes16-31 N=lane-16,M=8+r)
  const int rbase = (lane < 16) ? 0 : 8;
#pragma unroll
  for (int i = 0; i < 2; ++i)
#pragma unroll
    for (int j = 0; j < 4; ++j)
#pragma unroll
      for (int r = 0; r < 8; ++r) {
        int row = m0 + wm + i * 16 + r + rbase;
        int col = n0 + wn + j * 16 + (lane & 15);
        qkv[(size_t)row * K3_ + col] = (_Float16)acc[i][j][r];
      }
}

// =====================================================================
// Kernel 2: gates = sigmoid(x @ W_gates^T), split into i/f, (B,H,S) layout
// =====================================================================
__global__ __launch_bounds__(256) void gates_kernel(
    const float* __restrict__ x, const float* __restrict__ Wg,
    float* __restrict__ ig, float* __restrict__ fg) {
  const int tid = threadIdx.x;
  const int o   = tid & 15;              // output gate index 0..15
  const int sl  = tid >> 4;              // 0..15
  const int m   = blockIdx.x * 16 + sl;  // 0..4095
  const int b   = m >> 11;
  const int s   = m & (S_ - 1);

  const float* xr = x + (size_t)m * DM_;
  const float* wr = Wg + (size_t)o * DM_;
  float acc = 0.f;
  for (int k = 0; k < DM_; k += 4) {
    v4f xv = *(const v4f*)(xr + k);
    v4f wv = *(const v4f*)(wr + k);
    acc += xv[0] * wv[0] + xv[1] * wv[1] + xv[2] * wv[2] + xv[3] * wv[3];
  }
  const float g = 1.f / (1.f + __expf(-acc));
  if (o < H_) ig[(size_t)(b * H_ + o) * S_ + s] = g;
  else        fg[(size_t)(b * H_ + (o - H_)) * S_ + s] = g;
}

// =====================================================================
// Kernel 3: gated causal attention per (b,h); flash-style, WMMA QK^T & PV
// block = 128 queries, 8 waves x 16 queries; V staged via async LDS copy
// =====================================================================
__global__ __launch_bounds__(256) void attn_kernel(
    const _Float16* __restrict__ qkv, const float* __restrict__ ig,
    const float* __restrict__ fg, float* __restrict__ hout) {
  __shared__ _Float16 Kt[DH_][32];      // transposed key tile [d][t]
  __shared__ _Float16 Vt[32][DH_];      // value tile [t][d]
  __shared__ float    scL[8][16][32];   // per-wave raw scores
  __shared__ _Float16 pL[8][16][32];    // per-wave probabilities (f16)
  __shared__ float    alphaL[8][16];
  __shared__ float    lL[8][16];
  __shared__ float    igL[8][16];

  const int tid  = threadIdx.x;
  const int lane = tid & 31;
  const int w    = tid >> 5;
  const int bh   = blockIdx.y;
  const int b    = bh / H_;
  const int hh   = bh % H_;
  const int q0   = blockIdx.x * 128;
  const int qbase = q0 + w * 16;

  const _Float16* qptr = qkv + (size_t)b * S_ * K3_ + (size_t)hh * DH_;
  const _Float16* kptr = qptr + DI_;
  const _Float16* vptr = qptr + 2 * DI_;

  // resident Q fragments over DH=256 (8 chunks of K=32)
  v16h qf[8];
  {
    const int m  = lane & 15;
    const int kb = (lane >> 4) * 8;
    const _Float16* qr = qptr + (size_t)(qbase + m) * K3_;
#pragma unroll
    for (int c = 0; c < 8; ++c) qf[c] = load_afrag(qr + c * 32, kb);
  }

  float mrun = -1e30f, lrun = 0.f, fgv = 0.f, igv = 0.f;
  if (lane < 16) {
    const int q = qbase + lane;
    fgv = fg[(size_t)(b * H_ + hh) * S_ + q];
    igv = ig[(size_t)(b * H_ + hh) * S_ + q];
  }

  v8f accO[16];
#pragma unroll
  for (int j = 0; j < 16; ++j) accO[j] = zero8();

  const float sc_scale = 0.0625f;  // 1/sqrt(256)
  const int tEnd = q0 + 128;
  for (int t0 = 0; t0 < tEnd; t0 += 32) {
    __syncthreads();
    // stage K (transposed, manual) and V (async copy) tiles into LDS
    {
      const int kk = tid >> 3;         // key row 0..31
      const int d0 = (tid & 7) * 32;   // dh chunk
      const _Float16* kr = kptr + (size_t)(t0 + kk) * K3_ + d0;
      const _Float16* vr = vptr + (size_t)(t0 + kk) * K3_ + d0;
#pragma unroll
      for (int j = 0; j < 32; j += 8) {
        v8h kv = *(const v8h*)(kr + j);
#pragma unroll
        for (int e = 0; e < 8; ++e) Kt[d0 + j + e][kk] = kv[e];
#if USE_ASYNC_LDS
        __builtin_amdgcn_global_load_async_to_lds_b128(
            to_glob_v4(vr + j), to_lds_v4(&Vt[kk][d0 + j]), 0, 0);
#else
        *(v8h*)&Vt[kk][d0 + j] = *(const v8h*)(vr + j);
#endif
      }
      if (t0 + 32 < tEnd) {  // prime GL2 with next key tile
        __builtin_prefetch(kr + (size_t)32 * K3_, 0, 1);
        __builtin_prefetch(vr + (size_t)32 * K3_, 0, 1);
      }
    }
#if USE_ASYNC_LDS
    __builtin_amdgcn_s_wait_asynccnt(0);
#endif
    __syncthreads();

    // scores 16x32 = Q(16xDH) x K^T(DHx32), two N-tiles of 16
    v8f sc0 = zero8(), sc1 = zero8();
#pragma unroll
    for (int c = 0; c < 8; ++c) {
      v16h b0, b1;
#pragma unroll
      for (int s = 0; s < 16; ++s) {
        const int kk = c * 32 + (s < 8 ? s : s + 8) + (lane >> 4) * 8;
        b0[s] = Kt[kk][lane & 15];
        b1[s] = Kt[kk][16 + (lane & 15)];
      }
      sc0 = WMMA_F16(qf[c], b0, sc0);
      sc1 = WMMA_F16(qf[c], b1, sc1);
    }
    // spill scores to per-wave LDS
    {
      const int n = lane & 15;
      const int rb = (lane < 16) ? 0 : 8;
#pragma unroll
      for (int r = 0; r < 8; ++r) {
        scL[w][r + rb][n]      = sc0[r];
        scL[w][r + rb][16 + n] = sc1[r];
      }
    }
    __syncthreads();

    // online softmax update (lanes 0..15 own one query row each)
    if (lane < 16) {
      const int q = qbase + lane;
      float rm = -1e30f;
      for (int j = 0; j < 32; ++j) {
        float v = (t0 + j <= q) ? scL[w][lane][j] * sc_scale * fgv : -1e30f;
        scL[w][lane][j] = v;
        rm = fmaxf(rm, v);
      }
      const float mnew  = fmaxf(mrun, rm);
      const float alpha = __expf(mrun - mnew);
      float psum = 0.f;
      for (int j = 0; j < 32; ++j) {
        float v = scL[w][lane][j];
        float p = (v > -1e29f) ? __expf(v - mnew) : 0.f;
        pL[w][lane][j] = (_Float16)p;
        psum += p;
      }
      lrun = lrun * alpha + psum;
      mrun = mnew;
      alphaL[w][lane] = alpha;
    }
    __syncthreads();

    // rescale accumulators + PV WMMAs
    float al[8];
    {
      const int rb = (lane < 16) ? 0 : 8;
#pragma unroll
      for (int r = 0; r < 8; ++r) al[r] = alphaL[w][r + rb];
    }
    v16h pf = load_afrag(&pL[w][lane & 15][0], (lane >> 4) * 8);
#pragma unroll
    for (int j = 0; j < 16; ++j) {
#pragma unroll
      for (int r = 0; r < 8; ++r) accO[j][r] *= al[r];
      v16h bv;
#pragma unroll
      for (int s = 0; s < 16; ++s)
        bv[s] = Vt[(s < 8 ? s : s + 8) + (lane >> 4) * 8][j * 16 + (lane & 15)];
      accO[j] = WMMA_F16(pf, bv, accO[j]);
    }
  }

  // epilogue: h = (attn@V)/l * i_gate
  __syncthreads();
  if (lane < 16) {
    lL[w][lane]  = fmaxf(lrun, 1e-20f);
    igL[w][lane] = igv;
  }
  __syncthreads();
  {
    const int rb = (lane < 16) ? 0 : 8;
    const int n  = lane & 15;
#pragma unroll
    for (int r = 0; r < 8; ++r) {
      const int s = qbase + r + rb;
      const float inv = igL[w][r + rb] / lL[w][r + rb];
      float* hr = hout + ((size_t)b * S_ + s) * DI_ + hh * DH_;
#pragma unroll
      for (int j = 0; j < 16; ++j) hr[j * 16 + n] = accO[j][r] * inv;
    }
  }
}

// =====================================================================
// Kernel 4: GroupNorm stats per (b, head-group): mean & rstd over (S, DH)
// =====================================================================
__global__ __launch_bounds__(256) void gn_stats_kernel(
    const float* __restrict__ h, float* __restrict__ stats) {
  __shared__ float sbuf[2][256];
  const int tid = threadIdx.x;
  const int bh  = blockIdx.x;
  const int b   = bh / H_;
  const int g   = bh % H_;
  const float* base = h + (size_t)b * S_ * DI_ + g * DH_;

  float sum = 0.f, sq = 0.f;
  for (int s = 0; s < S_; ++s) {
    const float* row = base + (size_t)s * DI_;
    for (int d = tid; d < DH_; d += 256) {
      float v = row[d];
      sum += v;
      sq  += v * v;
    }
  }
  sbuf[0][tid] = sum;
  sbuf[1][tid] = sq;
  __syncthreads();
  for (int off = 128; off > 0; off >>= 1) {
    if (tid < off) {
      sbuf[0][tid] += sbuf[0][tid + off];
      sbuf[1][tid] += sbuf[1][tid + off];
    }
    __syncthreads();
  }
  if (tid == 0) {
    const float N   = (float)(S_ * DH_);
    const float mu  = sbuf[0][0] / N;
    const float var = sbuf[1][0] / N - mu * mu;
    stats[2 * bh]     = mu;
    stats[2 * bh + 1] = rsqrtf(var + 1e-5f);
  }
}

// =====================================================================
// Kernel 5: out = GroupNorm(h) @ W_out^T + skip*x  (M=4096,N=1024,K=2048)
// normalization fused into A-tile staging; skip fused into epilogue;
// double-buffered LDS as in kernel 1
// =====================================================================
__global__ __launch_bounds__(256) void gemm_out_kernel(
    const float* __restrict__ h, const float* __restrict__ Wout,
    const float* __restrict__ gnw, const float* __restrict__ gnb,
    const float* __restrict__ stats, const float* __restrict__ x,
    const float* __restrict__ skip, float* __restrict__ out) {
  __shared__ _Float16 As[2][128][32];
  __shared__ _Float16 Bs[2][32][128];

  const int tid  = threadIdx.x;
  const int lane = tid & 31;
  const int w    = tid >> 5;
  const int m0   = blockIdx.y * 128;
  const int n0   = blockIdx.x * 128;
  const int wm   = (w >> 1) * 32;
  const int wn   = (w & 1) * 64;

  v8f acc[2][4];
#pragma unroll
  for (int i = 0; i < 2; ++i)
#pragma unroll
    for (int j = 0; j < 4; ++j) acc[i][j] = zero8();

  const int ar = tid >> 1;
  const int ac = (tid & 1) * 16;
  const int bidx = (m0 + ar) >> 11;  // batch of this A row
  const float* gaBase = h    + (size_t)(m0 + ar) * DI_ + ac;
  const float* gbBase = Wout + (size_t)(n0 + ar) * DI_ + ac;

  v4f areg[4], breg[4];
  // staging helper (normalize A): stage tile `kt` into buffer `buf`
  auto stage = [&](int kt, int buf) {
#pragma unroll
    for (int j = 0; j < 4; ++j)
#pragma unroll
      for (int e = 0; e < 4; ++e) {
        const int k   = kt * 32 + ac + 4 * j + e;
        const int grp = k >> 8;  // DH_=256 channels per group
        const float mu = stats[2 * (bidx * H_ + grp)];
        const float rs = stats[2 * (bidx * H_ + grp) + 1];
        As[buf][ar][ac + 4 * j + e] =
            (_Float16)((areg[j][e] - mu) * rs * gnw[k] + gnb[k]);
        Bs[buf][ac + 4 * j + e][ar] = (_Float16)breg[j][e];
      }
  };

  // prologue
#pragma unroll
  for (int j = 0; j < 4; ++j) {
    areg[j] = *(const v4f*)(gaBase + 4 * j);
    breg[j] = *(const v4f*)(gbBase + 4 * j);
  }
  stage(0, 0);
  __syncthreads();

  const int NT = DI_ / 32;
  for (int kt = 0; kt < NT; ++kt) {
    const int p = kt & 1;
    const bool more = (kt + 1 < NT);
    if (more) {
      const float* ga = gaBase + (size_t)(kt + 1) * 32;
      const float* gb = gbBase + (size_t)(kt + 1) * 32;
#pragma unroll
      for (int j = 0; j < 4; ++j) {
        areg[j] = *(const v4f*)(ga + 4 * j);
        breg[j] = *(const v4f*)(gb + 4 * j);
      }
      if (kt + 2 < NT) {
        __builtin_prefetch(gaBase + (size_t)(kt + 2) * 32, 0, 1);
        __builtin_prefetch(gbBase + (size_t)(kt + 2) * 32, 0, 1);
      }
    }

    v16h a[2], bf[4];
#pragma unroll
    for (int i = 0; i < 2; ++i)
      a[i] = load_afrag(&As[p][wm + i * 16 + (lane & 15)][0], (lane >> 4) * 8);
#pragma unroll
    for (int j = 0; j < 4; ++j)
#pragma unroll
      for (int s = 0; s < 16; ++s)
        bf[j][s] = Bs[p][(s < 8 ? s : s + 8) + (lane >> 4) * 8][wn + j * 16 + (lane & 15)];
#pragma unroll
    for (int i = 0; i < 2; ++i)
#pragma unroll
      for (int j = 0; j < 4; ++j) acc[i][j] = WMMA_F16(a[i], bf[j], acc[i][j]);

    if (more) stage(kt + 1, p ^ 1);
    __syncthreads();
  }

  const float ss = skip[0];
  const int rbase = (lane < 16) ? 0 : 8;
#pragma unroll
  for (int i = 0; i < 2; ++i)
#pragma unroll
    for (int j = 0; j < 4; ++j)
#pragma unroll
      for (int r = 0; r < 8; ++r) {
        const int row = m0 + wm + i * 16 + r + rbase;
        const int col = n0 + wn + j * 16 + (lane & 15);
        out[(size_t)row * DM_ + col] =
            acc[i][j][r] + ss * x[(size_t)row * DM_ + col];
      }
}

// =====================================================================
// Launch
// =====================================================================
extern "C" void kernel_launch(void* const* d_in, const int* in_sizes, int n_in,
                              void* d_out, int out_size, void* d_ws,
                              size_t ws_size, hipStream_t stream) {
  const float* x       = (const float*)d_in[0];
  const float* W_qkv   = (const float*)d_in[1];
  const float* W_gates = (const float*)d_in[2];
  const float* gn_w    = (const float*)d_in[3];
  const float* gn_b    = (const float*)d_in[4];
  const float* W_out   = (const float*)d_in[5];
  const float* skip    = (const float*)d_in[6];
  float* out = (float*)d_out;

  char* ws = (char*)d_ws;
  const size_t off_qkv   = 0;
  const size_t off_ig    = off_qkv + (size_t)MTOT_ * K3_ * sizeof(_Float16);  // 48MB
  const size_t off_fg    = off_ig + (size_t)B_ * H_ * S_ * sizeof(float);
  const size_t off_h     = off_fg + (size_t)B_ * H_ * S_ * sizeof(float);
  const size_t off_stats = off_h + (size_t)MTOT_ * DI_ * sizeof(float);       // +32MB

  _Float16* qkv = (_Float16*)(ws + off_qkv);
  float* ig     = (float*)(ws + off_ig);
  float* fg     = (float*)(ws + off_fg);
  float* h      = (float*)(ws + off_h);
  float* stats  = (float*)(ws + off_stats);

  gemm_qkv_kernel<<<dim3(K3_ / 128, MTOT_ / 128), 256, 0, stream>>>(x, W_qkv, qkv);
  gates_kernel<<<dim3(MTOT_ / 16), 256, 0, stream>>>(x, W_gates, ig, fg);
  attn_kernel<<<dim3(S_ / 128, B_ * H_), 256, 0, stream>>>(qkv, ig, fg, h);
  gn_stats_kernel<<<dim3(B_ * H_), 256, 0, stream>>>(h, stats);
  gemm_out_kernel<<<dim3(DM_ / 128, MTOT_ / 128), 256, 0, stream>>>(
      h, W_out, gn_w, gn_b, stats, x, skip, out);
}
